// MultiHeadGAT_51041391346029
// MI455X (gfx1250) — compile-verified, matrix operand order checked by reading, not run on hardware
//
#include <hip/hip_runtime.h>
#include <hip/hip_bf16.h>

typedef __attribute__((ext_vector_type(16))) __bf16 v16bf;
typedef __attribute__((ext_vector_type(8)))  float  v8f;

// ---------------- generic helpers ----------------

__global__ void k_fill_f32(float* p, float v, int n) {
    int i = blockIdx.x * blockDim.x + threadIdx.x;
    if (i < n) p[i] = v;
}

__global__ void k_f32_to_bf16(const float* s, __bf16* d, int n) {
    int i = blockIdx.x * blockDim.x + threadIdx.x;
    if (i < n) d[i] = (__bf16)s[i];
}

__device__ __forceinline__ void atomicMaxF(float* addr, float v) {
    // monotone IEEE trick: non-negative floats -> signed int max,
    // negative floats -> unsigned int min. Valid with init = -3e38.
    int vi = __float_as_int(v);
    if (vi >= 0) atomicMax((int*)addr, vi);
    else         atomicMin((unsigned int*)addr, (unsigned int)vi);
}

// ---------------- pack weights into WMMA B-fragment order (fuses f32->bf16) -------------
// Packed layout: [kt][nt][lane][e] with 16 bf16 per lane (32 bytes, contiguous).
// Fragment mapping (wave32, 16-bit B 32x16): lane<16 -> N=lane, K elems 0..7 = k0..k0+7,
// elems 8..15 = k0+16..k0+23 ; lane>=16 -> N=lane-16, K shifted by 8.

__global__ void k_pack_b(const float* __restrict__ B, __bf16* __restrict__ Bp, int K) {
    const int NC = 256;
    int i = blockIdx.x * blockDim.x + threadIdx.x;      // K*256 elements
    if (i < K * NC) {
        int e    = i & 15;
        int lane = (i >> 4) & 31;
        int t    = i >> 9;            // tile id = kt*16 + nt   (512 elems / tile)
        int nt   = t & 15;
        int kt   = t >> 4;
        int half = lane >> 4, lc = lane & 15;
        int n = nt * 16 + lc;
        int k = (e < 8) ? (kt * 32 + (half ? 8 : 0) + e)
                        : (kt * 32 + (half ? 24 : 16) + (e - 8));
        Bp[i] = (__bf16)B[(size_t)k * NC + n];
    }
}

// ---------------- edge coefficient precompute: c[j][h] = sum_f We[j, h*64+f]*ae[h,f] ----

__global__ void k_edge_coef(const float* __restrict__ We, const float* __restrict__ ae,
                            float* __restrict__ c) {
    int t = threadIdx.x;
    if (t < 8) {
        int j = t >> 2, h = t & 3;
        float s = 0.f;
        for (int f = 0; f < 64; ++f) s += We[j * 256 + h * 64 + f] * ae[h * 64 + f];
        c[j * 4 + h] = s;   // c[0..3]=c0[h], c[4..7]=c1[h]
    }
}

// ---------------- self-loop attr: deg + sum of incoming edge_attr ----------------

__global__ void k_deg_loop(const int* __restrict__ ei, const float* __restrict__ ea,
                           float* __restrict__ deg, float* __restrict__ la, int E) {
    int e = blockIdx.x * blockDim.x + threadIdx.x;
    if (e < E) {
        int d = ei[E + e];
        atomicAdd(&deg[d], 1.0f);
        atomicAdd(&la[d * 2 + 0], ea[e * 2 + 0]);
        atomicAdd(&la[d * 2 + 1], ea[e * 2 + 1]);
    }
}

__global__ void k_loop_fin(float* __restrict__ la, const float* __restrict__ deg, int N) {
    int i = blockIdx.x * blockDim.x + threadIdx.x;
    if (i < N) {
        float d = fmaxf(deg[i], 1.0f);
        la[i * 2 + 0] /= d;
        la[i * 2 + 1] /= d;
    }
}

// ---------------- WMMA bf16 GEMM: C[M x 256] = A[M x K] * Bpacked, f32 accum -----------
// block = 256 threads = 8 waves; block tile = 32 rows x 256 cols.
// Each wave: 2x2 group of 16x16 tiles -> 4 WMMAs / K-step with A and B fragment reuse.

__global__ __launch_bounds__(256)
void k_gemm_bf16(const __bf16* __restrict__ A, const __bf16* __restrict__ Bp,
                 float* __restrict__ C, int M, int K) {
    __shared__ __bf16 As[32][36];            // 32x32 A tile, padded
    const int NC   = 256;
    const int m0   = blockIdx.x * 32;
    const int tid  = threadIdx.x;
    const int wave = tid >> 5;
    const int lane = tid & 31;
    const int half = lane >> 4;
    const int lc   = lane & 15;
    const int nt0  = wave * 2;               // n-tile pair for this wave

    v8f acc00 = {}, acc01 = {}, acc10 = {}, acc11 = {};
    const int KT = K >> 5;

    for (int kt = 0; kt < KT; ++kt) {
        const int k0 = kt * 32;
        // stage 32x32 A tile into LDS (4 consecutive bf16 / thread, 8B aligned)
        {
            int idx = tid * 4;
            int r = idx >> 5, cc = idx & 31;
            int gm = m0 + r;
            if (gm < M) {
                const __bf16* ap = A + (size_t)gm * K + k0 + cc;
                if (kt + 1 < KT) __builtin_prefetch(ap + 32, 0, 1);   // next K-slice
#pragma unroll
                for (int j = 0; j < 4; ++j) As[r][cc + j] = ap[j];
            } else {
#pragma unroll
                for (int j = 0; j < 4; ++j) As[r][cc + j] = (__bf16)0.0f;
            }
        }
        __syncthreads();

        // A fragments (ISA 16-bit A 16x32 layout, wave32)
        v16bf a0, a1;
        const int kaLo = half ? 8 : 0;
        const int kaHi = half ? 24 : 16;
#pragma unroll
        for (int e = 0; e < 8; ++e) {
            a0[e]     = As[lc][kaLo + e];
            a0[8 + e] = As[lc][kaHi + e];
            a1[e]     = As[16 + lc][kaLo + e];
            a1[8 + e] = As[16 + lc][kaHi + e];
        }
        // B fragments: one contiguous 32B vector load per lane (coalesced)
        const v16bf b0 = *(const v16bf*)(Bp + (((size_t)kt * 16 + nt0) * 32 + lane) * 16);
        const v16bf b1 = *(const v16bf*)(Bp + (((size_t)kt * 16 + nt0 + 1) * 32 + lane) * 16);

        acc00 = __builtin_amdgcn_wmma_f32_16x16x32_bf16(false, a0, false, b0, (short)0, acc00, false, false);
        acc01 = __builtin_amdgcn_wmma_f32_16x16x32_bf16(false, a0, false, b1, (short)0, acc01, false, false);
        acc10 = __builtin_amdgcn_wmma_f32_16x16x32_bf16(false, a1, false, b0, (short)0, acc10, false, false);
        acc11 = __builtin_amdgcn_wmma_f32_16x16x32_bf16(false, a1, false, b1, (short)0, acc11, false, false);
        __syncthreads();
    }

    // C/D layout: lane<16 -> M=r, N=lane ; lane>=16 -> M=8+r, N=lane-16
    const int n0 = nt0 * 16;
#pragma unroll
    for (int r = 0; r < 8; ++r) {
        int row = half ? (8 + r) : r;
        int gm0 = m0 + row;
        int gm1 = m0 + 16 + row;
        if (gm0 < M) {
            C[(size_t)gm0 * NC + n0 + lc]      = acc00[r];
            C[(size_t)gm0 * NC + n0 + 16 + lc] = acc01[r];
        }
        if (gm1 < M) {
            C[(size_t)gm1 * NC + n0 + lc]      = acc10[r];
            C[(size_t)gm1 * NC + n0 + 16 + lc] = acc11[r];
        }
    }
}

// ---------------- per-node attention terms: a_s[n,h], a_d[n,h] ----------------

__global__ void k_node_attn(const float* __restrict__ H, const float* __restrict__ aw_s,
                            const float* __restrict__ aw_d, float* __restrict__ a_s,
                            float* __restrict__ a_d, int N) {
    int i = blockIdx.x * blockDim.x + threadIdx.x;
    if (i < N * 4) {
        int n = i >> 2, h = i & 3;
        const float* hp = H + (size_t)n * 256 + h * 64;
        float s = 0.f, d = 0.f;
        for (int f = 0; f < 64; ++f) {
            float v = hp[f];
            s += v * aw_s[h * 64 + f];
            d += v * aw_d[h * 64 + f];
        }
        a_s[i] = s; a_d[i] = d;
    }
}

// ---------------- edge logits (leaky relu) + segment max over dst ----------------

__global__ void k_edge_logit(const int* __restrict__ ei, const float* __restrict__ ea,
                             const float* __restrict__ la, const float* __restrict__ c,
                             const float* __restrict__ a_s, const float* __restrict__ a_d,
                             float* __restrict__ logit, float* __restrict__ amax,
                             int E, int N) {
    int i = blockIdx.x * blockDim.x + threadIdx.x;
    int tot = (E + N) * 4;
    if (i < tot) {
        int e = i >> 2, h = i & 3;
        int s, d; float e0, e1;
        if (e < E) { s = ei[e]; d = ei[E + e]; e0 = ea[e * 2]; e1 = ea[e * 2 + 1]; }
        else       { s = d = e - E;            e0 = la[s * 2]; e1 = la[s * 2 + 1]; }
        float x = a_s[s * 4 + h] + a_d[d * 4 + h] + e0 * c[h] + e1 * c[4 + h];
        x = x > 0.f ? x : 0.2f * x;       // leaky_relu, slope 0.2
        logit[i] = x;
        atomicMaxF(&amax[d * 4 + h], x);
    }
}

// ---------------- exp(logit - max) + segment sum ----------------

__global__ void k_edge_exp(const int* __restrict__ ei, float* __restrict__ logit,
                           const float* __restrict__ amax, float* __restrict__ denom,
                           int E, int N) {
    int i = blockIdx.x * blockDim.x + threadIdx.x;
    int tot = (E + N) * 4;
    if (i < tot) {
        int e = i >> 2, h = i & 3;
        int d = (e < E) ? ei[E + e] : (e - E);
        float ex = __expf(logit[i] - amax[d * 4 + h]);
        logit[i] = ex;
        atomicAdd(&denom[d * 4 + h], ex);
    }
}

// ---------------- message passing: acc[dst,c] += H[src,c] * alpha ----------------

__global__ void k_msg_pass(const int* __restrict__ ei, const float* __restrict__ H,
                           const float* __restrict__ ex, const float* __restrict__ denom,
                           float* __restrict__ acc, int E, int N) {
    int i = blockIdx.x * blockDim.x + threadIdx.x;   // (E+N)*256 threads
    int tot = (E + N) * 256;
    if (i < tot) {
        int e = i >> 8, cch = i & 255, h = cch >> 6;
        int s, d;
        if (e < E) { s = ei[e]; d = ei[E + e]; } else { s = d = e - E; }
        float w = ex[e * 4 + h] / (denom[d * 4 + h] + 1e-16f);
        atomicAdd(&acc[(size_t)d * 256 + cch], H[(size_t)s * 256 + cch] * w);
    }
}

// ---------------- layer-1 finalize: ELU(acc + b1) -> bf16 ----------------

__global__ void k_fin1(const float* __restrict__ acc, const float* __restrict__ b1,
                       __bf16* __restrict__ hbf, int N) {
    int i = blockIdx.x * blockDim.x + threadIdx.x;
    if (i < N * 256) {
        float v = acc[i] + b1[i & 255];
        v = v > 0.f ? v : (__expf(v) - 1.0f);
        hbf[i] = (__bf16)v;
    }
}

// ---------------- layer-2 finalize: ELU(mean_h(acc) + b2) -> f32 [N,64] ----------------

__global__ void k_fin2(const float* __restrict__ acc, const float* __restrict__ b2,
                       float* __restrict__ h2, int N) {
    int i = blockIdx.x * blockDim.x + threadIdx.x;
    if (i < N * 64) {
        int n = i >> 6, f = i & 63;
        const float* p = acc + (size_t)n * 256;
        float v = 0.25f * (p[f] + p[64 + f] + p[128 + f] + p[192 + f]) + b2[f];
        v = v > 0.f ? v : (__expf(v) - 1.0f);
        h2[i] = v;
    }
}

// ---------------- output heads: wire/terminal/action logits ----------------

__global__ void k_heads(const float* __restrict__ h2,
                        const float* __restrict__ Ww, const float* __restrict__ bw,
                        const float* __restrict__ Wt, const float* __restrict__ bt,
                        const float* __restrict__ Wa, const float* __restrict__ ba,
                        float* __restrict__ out, int N) {
    int i = blockIdx.x * blockDim.x + threadIdx.x;
    if (i < N * 80) {
        int n = i / 80, j = i % 80;
        const float* W; const float* b; float* o; int nc;
        if (j < 50)      { W = Ww; b = bw; o = out;                      nc = 50; }
        else if (j < 70) { j -= 50; W = Wt; b = bt; o = out + (size_t)N * 50; nc = 20; }
        else             { j -= 70; W = Wa; b = ba; o = out + (size_t)N * 70; nc = 10; }
        float s = b[j];
        const float* hp = h2 + (size_t)n * 64;
        for (int f = 0; f < 64; ++f) s += hp[f] * W[f * nc + j];
        o[(size_t)n * nc + j] = s;
    }
}

// ---------------- launch ----------------

static inline dim3 gsz(long long n, int b) { return dim3((unsigned)((n + b - 1) / b)); }

extern "C" void kernel_launch(void* const* d_in, const int* in_sizes, int n_in,
                              void* d_out, int out_size, void* d_ws, size_t ws_size,
                              hipStream_t stream) {
    const float* x   = (const float*)d_in[0];
    const int*   ei  = (const int*)d_in[1];
    const float* ea  = (const float*)d_in[2];
    const float* W1  = (const float*)d_in[3];
    const float* We1 = (const float*)d_in[4];
    const float* as1 = (const float*)d_in[5];
    const float* ad1 = (const float*)d_in[6];
    const float* ae1 = (const float*)d_in[7];
    const float* b1  = (const float*)d_in[8];
    const float* W2  = (const float*)d_in[9];
    const float* We2 = (const float*)d_in[10];
    const float* as2 = (const float*)d_in[11];
    const float* ad2 = (const float*)d_in[12];
    const float* ae2 = (const float*)d_in[13];
    const float* b2  = (const float*)d_in[14];
    const float* Ww  = (const float*)d_in[15];
    const float* bw  = (const float*)d_in[16];
    const float* Wt  = (const float*)d_in[17];
    const float* bt  = (const float*)d_in[18];
    const float* Wa  = (const float*)d_in[19];
    const float* ba  = (const float*)d_in[20];

    const int N  = in_sizes[0] / 128;
    const int E  = in_sizes[1] / 2;
    const int E2 = E + N;

    // workspace layout
    char* ws = (char*)d_ws;
    size_t off = 0;
    auto alloc = [&](size_t bytes) -> char* {
        char* p = ws + off;
        off = (off + bytes + 255) & ~(size_t)255;
        return p;
    };
    __bf16* xbf   = (__bf16*)alloc((size_t)N * 128 * 2);
    __bf16* W1p   = (__bf16*)alloc(128 * 256 * 2);        // fragment-packed
    __bf16* W2p   = (__bf16*)alloc(256 * 256 * 2);        // fragment-packed
    float*  ce1   = (float*)alloc(8 * 4);
    float*  ce2   = (float*)alloc(8 * 4);
    float*  Hbuf  = (float*)alloc((size_t)N * 256 * 4);   // transformed node features
    float*  a_s   = (float*)alloc((size_t)N * 4 * 4);
    float*  a_d   = (float*)alloc((size_t)N * 4 * 4);
    float*  deg   = (float*)alloc((size_t)N * 4);
    float*  la    = (float*)alloc((size_t)N * 2 * 4);
    float*  logit = (float*)alloc((size_t)E2 * 4 * 4);    // logits, then exp in-place
    float*  amax  = (float*)alloc((size_t)N * 4 * 4);
    float*  denom = (float*)alloc((size_t)N * 4 * 4);
    float*  accum = (float*)alloc((size_t)N * 256 * 4);
    __bf16* h1bf  = (__bf16*)alloc((size_t)N * 256 * 2);
    float*  h2    = (float*)alloc((size_t)N * 64 * 4);
    (void)ws_size; (void)n_in; (void)out_size;

    const int B = 256;

    // --- prep: bf16 conversion of x, fragment-packed weights, edge coefs, self-loops ---
    k_f32_to_bf16<<<gsz((size_t)N * 128, B), B, 0, stream>>>(x, xbf, N * 128);
    k_pack_b<<<gsz(128 * 256, B), B, 0, stream>>>(W1, W1p, 128);
    k_pack_b<<<gsz(256 * 256, B), B, 0, stream>>>(W2, W2p, 256);
    k_edge_coef<<<1, 32, 0, stream>>>(We1, ae1, ce1);
    k_edge_coef<<<1, 32, 0, stream>>>(We2, ae2, ce2);
    k_fill_f32<<<gsz(N, B), B, 0, stream>>>(deg, 0.f, N);
    k_fill_f32<<<gsz(N * 2, B), B, 0, stream>>>(la, 0.f, N * 2);
    k_deg_loop<<<gsz(E, B), B, 0, stream>>>(ei, ea, deg, la, E);
    k_loop_fin<<<gsz(N, B), B, 0, stream>>>(la, deg, N);

    // ================= layer 1 =================
    k_fill_f32<<<gsz(N * 4, B), B, 0, stream>>>(amax, -3.0e38f, N * 4);
    k_fill_f32<<<gsz(N * 4, B), B, 0, stream>>>(denom, 0.f, N * 4);
    k_fill_f32<<<gsz((size_t)N * 256, B), B, 0, stream>>>(accum, 0.f, N * 256);

    k_gemm_bf16<<<gsz((N + 31) / 32, 1).x, 256, 0, stream>>>(xbf, W1p, Hbuf, N, 128);
    k_node_attn<<<gsz(N * 4, B), B, 0, stream>>>(Hbuf, as1, ad1, a_s, a_d, N);
    k_edge_logit<<<gsz((size_t)E2 * 4, B), B, 0, stream>>>(ei, ea, la, ce1, a_s, a_d, logit, amax, E, N);
    k_edge_exp<<<gsz((size_t)E2 * 4, B), B, 0, stream>>>(ei, logit, amax, denom, E, N);
    k_msg_pass<<<gsz((size_t)E2 * 256, B), B, 0, stream>>>(ei, Hbuf, logit, denom, accum, E, N);
    k_fin1<<<gsz((size_t)N * 256, B), B, 0, stream>>>(accum, b1, h1bf, N);

    // ================= layer 2 =================
    k_fill_f32<<<gsz(N * 4, B), B, 0, stream>>>(amax, -3.0e38f, N * 4);
    k_fill_f32<<<gsz(N * 4, B), B, 0, stream>>>(denom, 0.f, N * 4);
    k_fill_f32<<<gsz((size_t)N * 256, B), B, 0, stream>>>(accum, 0.f, N * 256);

    k_gemm_bf16<<<gsz((N + 31) / 32, 1).x, 256, 0, stream>>>(h1bf, W2p, Hbuf, N, 256);
    k_node_attn<<<gsz(N * 4, B), B, 0, stream>>>(Hbuf, as2, ad2, a_s, a_d, N);
    k_edge_logit<<<gsz((size_t)E2 * 4, B), B, 0, stream>>>(ei, ea, la, ce2, a_s, a_d, logit, amax, E, N);
    k_edge_exp<<<gsz((size_t)E2 * 4, B), B, 0, stream>>>(ei, logit, amax, denom, E, N);
    k_msg_pass<<<gsz((size_t)E2 * 256, B), B, 0, stream>>>(ei, Hbuf, logit, denom, accum, E, N);
    k_fin2<<<gsz((size_t)N * 64, B), B, 0, stream>>>(accum, b2, h2, N);

    // ================= output heads =================
    k_heads<<<gsz((size_t)N * 80, B), B, 0, stream>>>(h2, Ww, bw, Wt, bt, Wa, ba, (float*)d_out, N);
}
